// PathVGAE_34943853920683
// MI455X (gfx1250) — matrix operand order repaired and checked by reading, not compiled, not run on hardware
//
#include <hip/hip_runtime.h>
#include <hip/hip_bf16.h>

typedef __attribute__((ext_vector_type(16))) _Float16 v16h;
typedef __attribute__((ext_vector_type(8)))  float    v8f;

static __device__ __forceinline__ size_t gtid() {
  return (size_t)blockIdx.x * blockDim.x + threadIdx.x;
}

// ---------------- utility kernels ----------------

__global__ void zero_kernel(float* __restrict__ p, size_t n) {
  size_t i = gtid();
  if (i < n) p[i] = 0.f;
}

__global__ void zero2d_kernel(float* __restrict__ dst, int ldd, int dof, int nf, int M) {
  size_t i = gtid();
  if (i >= (size_t)M * nf) return;
  int n = (int)(i / nf), f = (int)(i % nf);
  dst[(size_t)n * ldd + dof + f] = 0.f;
}

__global__ void copy2d_kernel(const float* __restrict__ src, int lds, int so,
                              float* __restrict__ dst, int ldd, int dof, int nf, int M) {
  size_t i = gtid();
  if (i >= (size_t)M * nf) return;
  int n = (int)(i / nf), f = (int)(i % nf);
  dst[(size_t)n * ldd + dof + f] = src[(size_t)n * lds + so + f];
}

__global__ void degree_kernel(const int* __restrict__ row, const int* __restrict__ col,
                              float* __restrict__ degr, float* __restrict__ degc, int E) {
  size_t e = gtid();
  if (e >= (size_t)E) return;
  atomicAdd(&degr[row[e]], 1.f);
  atomicAdd(&degc[col[e]], 1.f);
}

__global__ void inv_clamp_kernel(const float* __restrict__ a, float* __restrict__ o, int n) {
  size_t i = gtid();
  if (i < (size_t)n) o[i] = 1.f / fmaxf(a[i], 1.f);
}

__global__ void scale_rows_kernel(float* __restrict__ buf, int ld, int nf,
                                  const float* __restrict__ s, int M) {
  size_t i = gtid();
  if (i >= (size_t)M * nf) return;
  int n = (int)(i / nf), f = (int)(i % nf);
  buf[(size_t)n * ld + f] *= s[n];
}

// dst[col[e], :] += src[row[e], :] * (rscale ? rscale[row[e]] : 1)
__global__ void scatter_kernel(const float* __restrict__ src, int lds, int so,
                               float* __restrict__ dst, int ldd, int dof,
                               const int* __restrict__ row, const int* __restrict__ col,
                               const float* __restrict__ rscale, int E, int nf4) {
  size_t i = gtid();
  if (i >= (size_t)E * nf4) return;
  int e = (int)(i / nf4);
  int f = (int)(i % nf4) * 4;
  int r = row[e], c = col[e];
  float s = rscale ? rscale[r] : 1.f;
  const float4 v = *(const float4*)(src + (size_t)r * lds + so + f);
  float* dp = dst + (size_t)c * ldd + dof + f;
  atomicAdd(dp + 0, v.x * s);
  atomicAdd(dp + 1, v.y * s);
  atomicAdd(dp + 2, v.z * s);
  atomicAdd(dp + 3, v.w * s);
}

// out1[n] = bias + dot(A[n,:K], w); optionally mirrored into out2[n,o2off]
__global__ void rowdot_kernel(const float* __restrict__ A, int lda,
                              const float* __restrict__ w, int K,
                              const float* __restrict__ bias,
                              float* __restrict__ out1,
                              float* __restrict__ out2, int ldO2, int o2off, int M) {
  size_t n = gtid();
  if (n >= (size_t)M) return;
  float s = bias[0];
  for (int k = 0; k < K; ++k) s += A[n * lda + k] * w[k];
  out1[n] = s;
  if (out2) out2[n * ldO2 + o2off] = s;
}

// ---------------- WMMA weight pre-swizzle ----------------
// CDNA5 wave32 B-matrix (16-bit, 32x16 per tile) lane layout:
//   lane t: column N = (t&15); half h holds K = (h&7) + ((t>>4)<<3) + ((h>>3)<<4) within the 32-K tile.
// Fragment memory order: ((kt*nNt + nt)*32 + lane)*16 + h  -> one v16h load per lane.
__global__ void prep_frag_kernel(const float* __restrict__ Wsrc, int Ksrc, int Nsrc,
                                 int nKt, int nNt, _Float16* __restrict__ frag) {
  size_t idx = gtid();
  size_t tot = (size_t)nKt * nNt * 512;
  if (idx >= tot) return;
  int h    = (int)(idx & 15);
  int lane = (int)((idx >> 4) & 31);
  int tile = (int)(idx >> 9);
  int nt = tile % nNt;
  int kt = tile / nNt;
  int k = kt * 32 + (h & 7) + ((lane >> 4) << 3) + ((h >> 3) << 4);
  int n = nt * 16 + (lane & 15);
  float v = (k < Ksrc && n < Nsrc) ? Wsrc[(size_t)k * Nsrc + n] : 0.f;
  frag[idx] = (_Float16)v;
}

// ---------------- fused WMMA GEMM with async-LDS weight staging ----------------
// All waves in a block consume the same weight fragment; stage it ONCE per block
// into LDS with gfx1250 global_load_async_to_lds_b128 (ASYNCcnt), then feed the
// WMMA pipe from ds_load_b128.
// C[M x (NNT*16)] = A[M x (nKt*32)] @ Wfrag ; epilogue:
//   y = acc + bias[c] (+ addBuf[r,c]) ; BN ; ReLU ; + res[r,c] ; store out1 (and out2)
template<int NNT>
__global__ void wmma_gemm_kernel(const float* __restrict__ A, int lda,
                                 const _Float16* __restrict__ frag,
                                 const float* __restrict__ bias,
                                 const float* __restrict__ addBuf, int ldAdd,
                                 const float* __restrict__ bnG, const float* __restrict__ bnB,
                                 const float* __restrict__ bnM, const float* __restrict__ bnV,
                                 const float* __restrict__ res, int ldRes,
                                 float* __restrict__ out1, int ldO1, int o1off,
                                 float* __restrict__ out2, int ldO2, int o2off,
                                 int M, int nKt, int relu) {
  extern __shared__ _Float16 lfrag[];

  // --- cooperative async copy: global fragment -> LDS (16B per lane per iter) ---
  {
    const int nchunks = nKt * NNT * 64;           // 16-byte chunks
    const unsigned lbase = (unsigned)(uintptr_t)lfrag;  // LDS-relative byte offset
    for (int c = (int)threadIdx.x; c < nchunks; c += (int)blockDim.x) {
      unsigned lo = lbase + (unsigned)c * 16u;
      const char* gp = (const char*)frag + (size_t)c * 16u;
      asm volatile("global_load_async_to_lds_b128 %0, %1, off"
                   :: "v"(lo), "v"(gp) : "memory");
    }
    asm volatile("s_wait_asynccnt 0x0" ::: "memory");
    __syncthreads();   // before any wave-level early exit
  }

  const int wave = (int)(((size_t)blockIdx.x * blockDim.x + threadIdx.x) >> 5);
  const int nRowTiles = (M + 15) >> 4;
  if (wave >= nRowTiles) return;   // wave-uniform: EXEC stays all-ones for WMMA

  const int t     = threadIdx.x & 31;
  const int hi    = t >> 4;              // lane half: 0 or 1
  const int mrow  = (wave << 4) + (t & 15);
  const int kbase = hi << 3;             // A: lanes 16-31 carry K+8 groups

  const v8f vzero = {0.f, 0.f, 0.f, 0.f, 0.f, 0.f, 0.f, 0.f};
  v8f acc[NNT];
#pragma unroll
  for (int i = 0; i < NNT; ++i) acc[i] = vzero;

  for (int kt = 0; kt < nKt; ++kt) {
    // A (16-bit 16x32): halves 0-7 = K kbase..kbase+7, halves 8-15 = K kbase+16..kbase+23
    const float* ap = A + (size_t)mrow * lda + kt * 32 + kbase;
    v16h a;
#pragma unroll
    for (int j = 0; j < 8; ++j) {
      a[j]     = (_Float16)ap[j];
      a[j + 8] = (_Float16)ap[j + 16];
    }
    const v16h* bp = (const v16h*)lfrag + (size_t)kt * NNT * 32 + t;
#pragma unroll
    for (int nt = 0; nt < NNT; ++nt) {
      v16h b = bp[nt * 32];
      acc[nt] = __builtin_amdgcn_wmma_f32_16x16x32_f16(
          false, a, false, b, (short)0, acc[nt], false, false);
    }
  }

  // C/D layout: lane t -> col (t&15); VGPR r -> row r + (t>>4)*8
  const int rbase = (wave << 4) + (hi << 3);
  const int cl = t & 15;
#pragma unroll
  for (int nt = 0; nt < NNT; ++nt) {
    const int c = nt * 16 + cl;
    float sc = 1.f, sh = 0.f;
    if (bnG) {
      const float inv = rsqrtf(bnV[c] + 1e-5f);
      sc = bnG[c] * inv;
      sh = bnB[c] - bnM[c] * sc;
    }
    const float bb = bias ? bias[c] : 0.f;
#pragma unroll
    for (int r = 0; r < 8; ++r) {
      const int rowg = rbase + r;
      if (rowg >= M) continue;
      float y = acc[nt][r] + bb;
      if (addBuf) y += addBuf[(size_t)rowg * ldAdd + c];
      if (bnG)    y = y * sc + sh;
      if (relu)   y = fmaxf(y, 0.f);
      if (res)    y += res[(size_t)rowg * ldRes + c];
      out1[(size_t)rowg * ldO1 + o1off + c] = y;
      if (out2) out2[(size_t)rowg * ldO2 + o2off + c] = y;
    }
  }
}

// ---------------- host orchestration ----------------

extern "C" void kernel_launch(void* const* d_in, const int* in_sizes, int n_in,
                              void* d_out, int out_size, void* d_ws, size_t ws_size,
                              hipStream_t stream) {
  (void)n_in; (void)out_size; (void)ws_size;
  const int N = in_sizes[0] / 64;     // 50000
  const int E = in_sizes[1] / 2;      // 800000
  const int L = 4, STEPS = 4;

  const float* x     = (const float*)d_in[0];
  const int*   erow  = (const int*)d_in[1];
  const int*   ecol  = erow + E;
  const float* W_in  = (const float*)d_in[2];
  const float* b_in  = (const float*)d_in[3];
  const float* sWl   = (const float*)d_in[4];
  const float* sbl   = (const float*)d_in[5];
  const float* sWr   = (const float*)d_in[6];
  const float* bnG   = (const float*)d_in[7];
  const float* bnB   = (const float*)d_in[8];
  const float* bnM   = (const float*)d_in[9];
  const float* bnV   = (const float*)d_in[10];
  const float* vWl   = (const float*)d_in[11];
  const float* vbl   = (const float*)d_in[12];
  const float* vWr   = (const float*)d_in[13];
  const float* vbnG  = (const float*)d_in[14];
  const float* vbnB  = (const float*)d_in[15];
  const float* vbnM  = (const float*)d_in[16];
  const float* vbnV  = (const float*)d_in[17];
  const float* W_mu  = (const float*)d_in[18];
  const float* b_mu  = (const float*)d_in[19];
  const float* W_lv  = (const float*)d_in[20];
  const float* b_lv  = (const float*)d_in[21];
  const float* rkW1  = (const float*)d_in[22];
  const float* rkb1  = (const float*)d_in[23];
  const float* rkW2  = (const float*)d_in[24];
  const float* rkb2  = (const float*)d_in[25];
  const float* rgW1  = (const float*)d_in[26];
  const float* rgb1  = (const float*)d_in[27];
  const float* rgW2  = (const float*)d_in[28];
  const float* rgb2  = (const float*)d_in[29];
  const float* rgW3  = (const float*)d_in[30];
  const float* rgb3  = (const float*)d_in[31];

  float* out   = (float*)d_out;
  float* preds = out;
  float* rank  = out + N;
  float* mu    = out + 2 * (size_t)N;
  float* lv    = mu + (size_t)N * 64;

  float* w = (float*)d_ws;
  size_t o = 0;
  auto take = [&](size_t n) { float* p = w + o; o += n; return p; };
  float* h0   = take((size_t)N * 64);
  float* h1   = take((size_t)N * 64);
  float* tmp  = take((size_t)N * 64);
  float* agg  = take((size_t)N * 384);
  float* vgin = take((size_t)N * 384);   // stage1 cols 0..319, path cols 320..383
  float* rgin = take((size_t)N * 416);   // reg input, K padded 385 -> 416
  float* vbuf = take((size_t)N * 64);
  float* p0   = take((size_t)N * 64);
  float* p1   = take((size_t)N * 64);
  float* cntc = take(N);
  float* icnt = take(N);
  float* degr = take(N);
  float* ideg = take(N);

  _Float16* frag = (_Float16*)(w + o);
  size_t fo = 0;
  auto takeh = [&](size_t n) { _Float16* p = frag + fo; fo += n; return p; };
  _Float16* fWin = takeh(4096);
  _Float16* fWl[4]; for (int i = 0; i < 4; ++i) fWl[i] = takeh(4096);
  _Float16* fWr[4]; for (int i = 0; i < 4; ++i) fWr[i] = takeh(4096);
  _Float16* fvWl = takeh(24576);
  _Float16* fvWr = takeh(24576);
  _Float16* fWmu = takeh(4096);
  _Float16* fWlv = takeh(4096);
  _Float16* fR1  = takeh(4096);
  _Float16* fG1  = takeh((size_t)13 * 4 * 512);
  _Float16* fG2  = takeh((size_t)2 * 2 * 512);

  const int B = 256;
  auto blk = [](size_t n, int b) { return (unsigned)((n + (size_t)b - 1) / b); };
  const unsigned gemmGrid = blk((size_t)((N + 15) / 16) * 32, B);
  auto ldsBytes = [](int nKt, int nNt) { return (size_t)nKt * nNt * 1024; };

  // --- weight fragment prep (f32 -> f16, WMMA B layout) ---
  auto prep = [&](const float* src, int K, int Nn, int nkt, int nnt, _Float16* dst) {
    prep_frag_kernel<<<blk((size_t)nkt * nnt * 512, B), B, 0, stream>>>(src, K, Nn, nkt, nnt, dst);
  };
  prep(W_in, 64, 64, 2, 4, fWin);
  for (int i = 0; i < 4; ++i) prep(sWl + (size_t)i * 4096, 64, 64, 2, 4, fWl[i]);
  for (int i = 0; i < 4; ++i) prep(sWr + (size_t)i * 4096, 64, 64, 2, 4, fWr[i]);
  prep(vWl, 384, 64, 12, 4, fvWl);
  prep(vWr, 384, 64, 12, 4, fvWr);
  prep(W_mu, 64, 64, 2, 4, fWmu);
  prep(W_lv, 64, 64, 2, 4, fWlv);
  prep(rkW1, 64, 64, 2, 4, fR1);
  prep(rgW1, 385, 64, 13, 4, fG1);
  prep(rgW2, 64, 32, 2, 2, fG2);

  // --- degrees ---
  zero_kernel<<<blk(N, B), B, 0, stream>>>(cntc, (size_t)N);
  zero_kernel<<<blk(N, B), B, 0, stream>>>(degr, (size_t)N);
  degree_kernel<<<blk(E, B), B, 0, stream>>>(erow, ecol, degr, cntc, E);
  inv_clamp_kernel<<<blk(N, B), B, 0, stream>>>(cntc, icnt, N);
  inv_clamp_kernel<<<blk(N, B), B, 0, stream>>>(degr, ideg, N);

  // --- input skip: vgin[:,256:320] = x @ W_in + b_in ---
  wmma_gemm_kernel<4><<<gemmGrid, B, ldsBytes(2, 4), stream>>>(
      x, 64, fWin, b_in, nullptr, 0, nullptr, nullptr, nullptr, nullptr, nullptr, 0,
      vgin, 384, 256, nullptr, 0, 0, N, 2, 0);

  // --- 4 SAGE layers ---
  const float* hcur = x;
  float* hbufs[2] = {h0, h1};
  for (int i = 0; i < L; ++i) {
    float* hnext = hbufs[i & 1];
    zero_kernel<<<blk((size_t)N * 64, B), B, 0, stream>>>(agg, (size_t)N * 64);
    scatter_kernel<<<blk((size_t)E * 16, B), B, 0, stream>>>(
        hcur, 64, 0, agg, 64, 0, erow, ecol, nullptr, E, 16);
    scale_rows_kernel<<<blk((size_t)N * 64, B), B, 0, stream>>>(agg, 64, 64, icnt, N);
    // tmp = mean @ Wl + bl
    wmma_gemm_kernel<4><<<gemmGrid, B, ldsBytes(2, 4), stream>>>(
        agg, 64, fWl[i], sbl + i * 64, nullptr, 0, nullptr, nullptr, nullptr, nullptr,
        nullptr, 0, tmp, 64, 0, nullptr, 0, 0, N, 2, 0);
    // hnext = relu(bn(tmp + h @ Wr)) + h ; also into stage1 slot i
    wmma_gemm_kernel<4><<<gemmGrid, B, ldsBytes(2, 4), stream>>>(
        hcur, 64, fWr[i], nullptr, tmp, 64,
        bnG + i * 64, bnB + i * 64, bnM + i * 64, bnV + i * 64,
        hcur, 64, hnext, 64, 0, vgin, 384, i * 64, N, 2, 1);
    hcur = hnext;
  }

  // --- path aggregation: 4 normalized scatter steps; last lands in vgin[:,320:384] ---
  const float* pcur = hcur;
  float* pb[2] = {p0, p1};
  for (int s = 0; s < STEPS; ++s) {
    if (s < STEPS - 1) {
      float* d = pb[s & 1];
      zero_kernel<<<blk((size_t)N * 64, B), B, 0, stream>>>(d, (size_t)N * 64);
      scatter_kernel<<<blk((size_t)E * 16, B), B, 0, stream>>>(
          pcur, 64, 0, d, 64, 0, erow, ecol, ideg, E, 16);
      pcur = d;
    } else {
      zero2d_kernel<<<blk((size_t)N * 64, B), B, 0, stream>>>(vgin, 384, 320, 64, N);
      scatter_kernel<<<blk((size_t)E * 16, B), B, 0, stream>>>(
          pcur, 64, 0, vgin, 384, 320, erow, ecol, ideg, E, 16);
    }
  }

  // --- VGAE SAGE on 384-dim input ---
  zero_kernel<<<blk((size_t)N * 384, B), B, 0, stream>>>(agg, (size_t)N * 384);
  scatter_kernel<<<blk((size_t)E * 96, B), B, 0, stream>>>(
      vgin, 384, 0, agg, 384, 0, erow, ecol, nullptr, E, 96);
  scale_rows_kernel<<<blk((size_t)N * 384, B), B, 0, stream>>>(agg, 384, 384, icnt, N);
  wmma_gemm_kernel<4><<<gemmGrid, B, ldsBytes(12, 4), stream>>>(
      agg, 384, fvWl, vbl, nullptr, 0, nullptr, nullptr, nullptr, nullptr,
      nullptr, 0, tmp, 64, 0, nullptr, 0, 0, N, 12, 0);
  wmma_gemm_kernel<4><<<gemmGrid, B, ldsBytes(12, 4), stream>>>(
      vgin, 384, fvWr, nullptr, tmp, 64, vbnG, vbnB, vbnM, vbnV,
      nullptr, 0, vbuf, 64, 0, nullptr, 0, 0, N, 12, 1);

  // --- mu / logvar (mu mirrored into reg input cols 320..383) ---
  wmma_gemm_kernel<4><<<gemmGrid, B, ldsBytes(2, 4), stream>>>(
      vbuf, 64, fWmu, b_mu, nullptr, 0, nullptr, nullptr, nullptr, nullptr,
      nullptr, 0, mu, 64, 0, rgin, 416, 320, N, 2, 0);
  wmma_gemm_kernel<4><<<gemmGrid, B, ldsBytes(2, 4), stream>>>(
      vbuf, 64, fWlv, b_lv, nullptr, 0, nullptr, nullptr, nullptr, nullptr,
      nullptr, 0, lv, 64, 0, nullptr, 0, 0, N, 2, 0);

  // --- rank head: relu(z @ rkW1 + b) @ rkW2 + b ---
  wmma_gemm_kernel<4><<<gemmGrid, B, ldsBytes(2, 4), stream>>>(
      mu, 64, fR1, rkb1, nullptr, 0, nullptr, nullptr, nullptr, nullptr,
      nullptr, 0, p1, 64, 0, nullptr, 0, 0, N, 2, 1);
  rowdot_kernel<<<blk(N, B), B, 0, stream>>>(p1, 64, rkW2, 64, rkb2,
                                             rank, rgin, 416, 384, N);

  // --- regressor: reg_in = [stage1 | z | rank | pad0] (416) ---
  copy2d_kernel<<<blk((size_t)N * 320, B), B, 0, stream>>>(
      vgin, 384, 0, rgin, 416, 0, 320, N);
  zero2d_kernel<<<blk((size_t)N * 31, B), B, 0, stream>>>(rgin, 416, 385, 31, N);
  wmma_gemm_kernel<4><<<gemmGrid, B, ldsBytes(13, 4), stream>>>(
      rgin, 416, fG1, rgb1, nullptr, 0, nullptr, nullptr, nullptr, nullptr,
      nullptr, 0, tmp, 64, 0, nullptr, 0, 0, N, 13, 1);
  wmma_gemm_kernel<2><<<gemmGrid, B, ldsBytes(2, 2), stream>>>(
      tmp, 64, fG2, rgb2, nullptr, 0, nullptr, nullptr, nullptr, nullptr,
      nullptr, 0, p0, 32, 0, nullptr, 0, 0, N, 2, 1);
  rowdot_kernel<<<blk(N, B), B, 0, stream>>>(p0, 32, rgW3, 32, rgb3,
                                             preds, nullptr, 0, 0, N);
}
